// MemoryBlock_26439818674341
// MI455X (gfx1250) — matrix operand order repaired and checked by reading, not compiled, not run on hardware
//
#include <hip/hip_runtime.h>

// ---------------------------------------------------------------------------
// MemoryBlock LSTM for MI455X (gfx1250, wave32, WMMA).
// gates = [x ; h_prev] @ [W_ih ; W_hh]^T + b, fused per-step kernel.
// Precision: bf16 WMMA (16x16x32) with fp32 accumulate (reference is fp32;
// bf16 matrix path is the only way the 550 GFLOP of GEMM stays near the
// 23.3 TB/s memory roofline on this chip).
// ---------------------------------------------------------------------------

#define HDIM  512          // LSTM hidden
#define NGATE 2048         // 4*H
#define MROWS 8192         // nt*na rows per step
#define SEQ   16
#define KTILE 32           // K per WMMA
#define RT    64           // rows per workgroup tile
#define NT    32           // hidden cols per workgroup tile

typedef __attribute__((ext_vector_type(16))) __bf16 v16bf;
typedef __attribute__((ext_vector_type(8)))  float  v8f;

union FragBF { v16bf v; uint4 u[2]; };

__device__ __forceinline__ unsigned int bf16_1(float f) {
  unsigned int u = __float_as_uint(f);
  return (u + 0x7FFFu + ((u >> 16) & 1u)) >> 16;   // round-to-nearest-even
}
__device__ __forceinline__ unsigned int bf16_2(float lo, float hi) {
  return bf16_1(lo) | (bf16_1(hi) << 16);
}
__device__ __forceinline__ float fsigmoid(float x) {
  return 1.0f / (1.0f + __expf(-x));
}
__device__ __forceinline__ float ftanh(float x) {
  // saturates correctly at +-1 for |x| large
  return 1.0f - 2.0f / (__expf(2.0f * x) + 1.0f);
}

// --------------------------- prep kernels ----------------------------------

__global__ __launch_bounds__(256)
void prep_weights(const float* __restrict__ Wih, const float* __restrict__ Whh,
                  const float* __restrict__ b_ih, const float* __restrict__ b_hh,
                  unsigned short* __restrict__ WihB, unsigned short* __restrict__ WhhB,
                  float* __restrict__ bias)
{
  int i = blockIdx.x * blockDim.x + threadIdx.x;       // float4 index
  float4 a = ((const float4*)Wih)[i];
  float4 b = ((const float4*)Whh)[i];
  uint2 ua, ub;
  ua.x = bf16_2(a.x, a.y); ua.y = bf16_2(a.z, a.w);
  ub.x = bf16_2(b.x, b.y); ub.y = bf16_2(b.z, b.w);
  ((uint2*)WihB)[i] = ua;
  ((uint2*)WhhB)[i] = ub;
  if (i < NGATE) bias[i] = b_ih[i] + b_hh[i];
}

__global__ __launch_bounds__(256)
void prep_state(const float* __restrict__ hxs, const float* __restrict__ cxs,
                unsigned short* __restrict__ h0b, float* __restrict__ c0)
{
  int f = blockIdx.x * blockDim.x + threadIdx.x;       // float4 idx over 8192x512
  int r  = f >> 7;                                     // row (512/4 = 128 f4/row)
  int c4 = f & 127;
  int t = r >> 7, a = r & 127;                         // r = t*128 + a
  // hxs[::16] -> batch index 16*t ; hxs is (bs, na, H)
  size_t srcf4 = ((((size_t)t * SEQ) * 128 + a) * 512 + (size_t)c4 * 4) >> 2;
  float4 h = ((const float4*)hxs)[srcf4];
  float4 c = ((const float4*)cxs)[srcf4];
  uint2 hb; hb.x = bf16_2(h.x, h.y); hb.y = bf16_2(h.z, h.w);
  ((uint2*)h0b)[f] = hb;
  ((float4*)c0)[f] = c;
}

// --------------------------- fused step kernel -----------------------------
// grid = (MROWS/RT, HDIM/NT), block = 128 (4 waves).
// Wave w computes gate w's 64x32 fp32 tile via 16x16x32 bf16 WMMA, K = 1024
// (phase 0: x features vs W_ih, phase 1: masked h_prev vs W_hh).

__global__ __launch_bounds__(128)
void lstm_step(const float* __restrict__ h_self, const float* __restrict__ h_inter,
               const int* __restrict__ reset,
               const unsigned short* __restrict__ Wih, const unsigned short* __restrict__ Whh,
               const float* __restrict__ bias,
               const unsigned short* __restrict__ hin, unsigned short* __restrict__ hout,
               const float* __restrict__ cprev,
               float* __restrict__ outh, float* __restrict__ outc, int s)
{
  __shared__ uint4 ldsA[RT * 5];           // 64 rows x 80B (padded: conflict-free b128)
  __shared__ float ldsG[4][RT][NT];        // per-gate result tiles (32 KB)

  const int tid  = threadIdx.x;
  const int lane = tid & 31;
  const int wave = tid >> 5;               // = gate index 0..3
  const int half = lane >> 4;
  const int ln   = lane & 15;

  const int row_base = blockIdx.x * RT;
  const int col_base = blockIdx.y * NT;

  v8f acc[4][2] = {};                      // 4 row-subtiles x 2 col-subtiles

  // staging / epilogue mapping: 2 threads per row
  const int rr   = tid >> 1;               // 0..63
  const int part = tid & 1;                // 16-col half
  const int row_g = row_base + rr;
  const int t_st  = row_g >> 7;            // trajectory
  const int a_st  = row_g & 127;           // agent
  const int keep  = 1 - reset[(t_st << 4) + s];

  for (int phase = 0; phase < 2; ++phase) {
    const unsigned short* __restrict__ Wb = phase ? Whh : Wih;
    for (int kt = 0; kt < HDIM; kt += KTILE) {
      __syncthreads();
      if (phase == 0) {
        // x = concat(h_self[..., 256:], h_inter) at batch b = t*16 + s
        const float* src;
        if (kt < 256)
          src = h_self  + ((size_t)((t_st * SEQ + s) * 128 + a_st)) * 512
                        + (256 + kt + part * 16);
        else
          src = h_inter + ((size_t)((t_st * SEQ + s) * 128 + a_st)) * 256
                        + (kt - 256 + part * 16);
        const float4* s4 = (const float4*)src;
        float4 f0 = s4[0], f1 = s4[1], f2 = s4[2], f3 = s4[3];
        uint4 u0, u1;
        u0.x = bf16_2(f0.x, f0.y); u0.y = bf16_2(f0.z, f0.w);
        u0.z = bf16_2(f1.x, f1.y); u0.w = bf16_2(f1.z, f1.w);
        u1.x = bf16_2(f2.x, f2.y); u1.y = bf16_2(f2.z, f2.w);
        u1.z = bf16_2(f3.x, f3.y); u1.w = bf16_2(f3.z, f3.w);
        ldsA[rr * 5 + part * 2 + 0] = u0;
        ldsA[rr * 5 + part * 2 + 1] = u1;
      } else {
        // h_prev (already bf16), reset mask zeroes whole rows
        const uint4* s4 = (const uint4*)(hin + (size_t)row_g * HDIM + kt + part * 16);
        uint4 u0 = s4[0], u1 = s4[1];
        if (!keep) { u0 = make_uint4(0,0,0,0); u1 = make_uint4(0,0,0,0); }
        ldsA[rr * 5 + part * 2 + 0] = u0;
        ldsA[rr * 5 + part * 2 + 1] = u1;
      }
      __syncthreads();

      // B fragments straight from L2-resident bf16 weights.
      // B is KxN with B[k][n] = W[n][k]; lane holds N = ln, K = kt+half*16+j,
      // i.e. 32 contiguous bytes of W's row n.
      FragBF bf0, bf1;
      {
        const int n0 = wave * HDIM + col_base + ln;
        const uint4* g0 = (const uint4*)(Wb + (size_t)n0 * HDIM + kt + half * 16);
        bf0.u[0] = g0[0]; bf0.u[1] = g0[1];
        const uint4* g1 = (const uint4*)(Wb + (size_t)(n0 + 16) * HDIM + kt + half * 16);
        bf1.u[0] = g1[0]; bf1.u[1] = g1[1];
      }
#pragma unroll
      for (int rs = 0; rs < 4; ++rs) {
        // A fragment: row M = ln, chunk0 = K[half*16 .. +7], chunk1 = K[half*16+8 .. +7]
        FragBF af;
        const int abase = (rs * 16 + ln) * 5 + half * 2;
        af.u[0] = ldsA[abase];
        af.u[1] = ldsA[abase + 1];
        acc[rs][0] = __builtin_amdgcn_wmma_f32_16x16x32_bf16(
            false, af.v, false, bf0.v, (short)0, acc[rs][0], false, false);
        acc[rs][1] = __builtin_amdgcn_wmma_f32_16x16x32_bf16(
            false, af.v, false, bf1.v, (short)0, acc[rs][1], false, false);
      }
    }
  }

  // exchange gate tiles across waves
  __syncthreads();
#pragma unroll
  for (int rs = 0; rs < 4; ++rs)
#pragma unroll
    for (int cs = 0; cs < 2; ++cs)
#pragma unroll
      for (int v = 0; v < 8; ++v)
        ldsG[wave][rs * 16 + half * 8 + v][cs * 16 + ln] = acc[rs][cs][v];
  __syncthreads();

  // fused LSTM cell: thread handles row rr, 16 hidden cols
  const float km = (float)keep;
  const int hcol0 = col_base + part * 16;
  const size_t orow = (size_t)row_g * HDIM + hcol0;
#pragma unroll
  for (int q = 0; q < 4; ++q) {
    const int lc = part * 16 + q * 4;
    float4 iv = *(const float4*)&ldsG[0][rr][lc];
    float4 fv = *(const float4*)&ldsG[1][rr][lc];
    float4 gv = *(const float4*)&ldsG[2][rr][lc];
    float4 ov = *(const float4*)&ldsG[3][rr][lc];
    float4 bi  = *(const float4*)&bias[0 * HDIM + hcol0 + q * 4];
    float4 bff = *(const float4*)&bias[1 * HDIM + hcol0 + q * 4];
    float4 bg  = *(const float4*)&bias[2 * HDIM + hcol0 + q * 4];
    float4 bo  = *(const float4*)&bias[3 * HDIM + hcol0 + q * 4];
    float4 cp  = *(const float4*)&cprev[orow + q * 4];
    const float* ip  = (const float*)&iv;  const float* fp  = (const float*)&fv;
    const float* gp  = (const float*)&gv;  const float* op  = (const float*)&ov;
    const float* bip = (const float*)&bi;  const float* bfp = (const float*)&bff;
    const float* bgp = (const float*)&bg;  const float* bop = (const float*)&bo;
    const float* cpp = (const float*)&cp;
    float cn[4], hn[4];
#pragma unroll
    for (int j = 0; j < 4; ++j) {
      float ig = fsigmoid(ip[j] + bip[j]);
      float fg = fsigmoid(fp[j] + bfp[j]);
      float gg = ftanh(gp[j] + bgp[j]);
      float og = fsigmoid(op[j] + bop[j]);
      float c  = fg * (cpp[j] * km) + ig * gg;
      cn[j] = c;
      hn[j] = og * ftanh(c);
    }
    *(float4*)&outc[orow + q * 4] = make_float4(cn[0], cn[1], cn[2], cn[3]);
    *(float4*)&outh[orow + q * 4] = make_float4(hn[0], hn[1], hn[2], hn[3]);
    uint2 hb; hb.x = bf16_2(hn[0], hn[1]); hb.y = bf16_2(hn[2], hn[3]);
    *(uint2*)&hout[orow + q * 4] = hb;
  }
}

// ------------------------------- launcher ----------------------------------

extern "C" void kernel_launch(void* const* d_in, const int* in_sizes, int n_in,
                              void* d_out, int out_size, void* d_ws, size_t ws_size,
                              hipStream_t stream)
{
  (void)in_sizes; (void)n_in; (void)out_size; (void)ws_size;
  // inputs: 0:x 1:hxs 2:cxs 3:h_self 4:h_inter 5:reset 6:W_ih 7:W_hh 8:b_ih 9:b_hh
  const float* hxs     = (const float*)d_in[1];
  const float* cxs     = (const float*)d_in[2];
  const float* h_self  = (const float*)d_in[3];
  const float* h_inter = (const float*)d_in[4];
  const int*   reset   = (const int*)d_in[5];
  const float* W_ih    = (const float*)d_in[6];
  const float* W_hh    = (const float*)d_in[7];
  const float* b_ih    = (const float*)d_in[8];
  const float* b_hh    = (const float*)d_in[9];

  char* ws = (char*)d_ws;
  auto alignup = [](size_t x) { return (x + 255) & ~(size_t)255; };
  size_t off = 0;
  unsigned short* WihB = (unsigned short*)(ws + off); off = alignup(off + (size_t)NGATE * HDIM * 2);
  unsigned short* WhhB = (unsigned short*)(ws + off); off = alignup(off + (size_t)NGATE * HDIM * 2);
  float*          bias = (float*)(ws + off);          off = alignup(off + (size_t)NGATE * 4);
  float*          c0   = (float*)(ws + off);          off = alignup(off + (size_t)MROWS * HDIM * 4);
  unsigned short* hA   = (unsigned short*)(ws + off); off = alignup(off + (size_t)MROWS * HDIM * 2);
  unsigned short* hB   = (unsigned short*)(ws + off); off = alignup(off + (size_t)MROWS * HDIM * 2);
  // total ws use ~38 MB

  prep_weights<<<(NGATE * HDIM / 4) / 256, 256, 0, stream>>>(
      W_ih, W_hh, b_ih, b_hh, WihB, WhhB, bias);
  prep_state<<<(MROWS * HDIM / 4) / 256, 256, 0, stream>>>(hxs, cxs, hA, c0);

  float* outh = (float*)d_out;                         // hys: (16, 8192, 512) row-major
  float* outc = outh + (size_t)SEQ * MROWS * HDIM;     // cys: same layout
  for (int s = 0; s < SEQ; ++s) {
    const unsigned short* hin = (s & 1) ? hB : hA;
    unsigned short*       hot = (s & 1) ? hA : hB;
    const float* cp = (s == 0) ? c0 : (outc + (size_t)(s - 1) * MROWS * HDIM);
    lstm_step<<<dim3(MROWS / RT, HDIM / NT), 128, 0, stream>>>(
        h_self, h_inter, reset, WihB, WhhB, bias, hin, hot, cp,
        outh + (size_t)s * MROWS * HDIM, outc + (size_t)s * MROWS * HDIM, s);
  }
}
